// GGNN_10969346474844
// MI455X (gfx1250) — matrix operand order repaired and checked by reading, not compiled, not run on hardware
//
#include <hip/hip_runtime.h>
#include <hip/hip_bf16.h>

// ---------------------------------------------------------------------------
// GGNN forward, MI455X (gfx1250): wave32, v_wmma_f32_16x16x32_f16 for all
// GEMMs, TDM (tensor_load_to_lds) staging of the aggregation B-operand.
// B=64, N=512, D=64, E=4, A_DIM=16, STEPS=5
// ---------------------------------------------------------------------------

typedef __attribute__((ext_vector_type(16))) _Float16 v16h;
typedef __attribute__((ext_vector_type(8)))  _Float16 v8h;
typedef __attribute__((ext_vector_type(8)))  float    v8f;
typedef __attribute__((ext_vector_type(4)))  unsigned int u32x4;
typedef __attribute__((ext_vector_type(8)))  int          i32x8;
typedef __attribute__((ext_vector_type(4)))  int          i32x4;

#define WMMA_F16(a, b, c) \
  __builtin_amdgcn_wmma_f32_16x16x32_f16(false, (a), false, (b), (short)0, (c), false, false)

constexpr int B_  = 64;
constexpr int N_  = 512;
constexpr int D_  = 64;
constexpr int E_  = 4;
constexpr int AD_ = 16;
constexpr int BN_ = B_ * N_;   // 32768

constexpr int LDB = 136;       // LDS leading dim (halves): 128 + 8 pad -> bank-spread

// ---------------- fragment loaders (gfx1250 wave32 WMMA layouts) ------------
// A-matrix 16x32 f16 (M x K), row-major source, leading dim ld (elements).
// ISA 7.12.2: lane<16 -> row=lane, K in {0..7,16..23}; lane>=16 -> row=lane-16,
// K in {8..15,24..31}.  Two 16B contiguous loads per lane.
__device__ __forceinline__ v16h frag_a_f16(const _Float16* tile, int ld) {
  const int lane = threadIdx.x & 31;
  const int row  = lane & 15;
  const int kb   = (lane >> 4) << 3;           // 0 or 8
  const v8h lo = *reinterpret_cast<const v8h*>(tile + row * ld + kb);
  const v8h hi = *reinterpret_cast<const v8h*>(tile + row * ld + kb + 16);
  v16h f;
#pragma unroll
  for (int i = 0; i < 8; ++i) { f[i] = lo[i]; f[i + 8] = hi[i]; }
  return f;
}

// Same A layout built from an fp32 row-major source (convert on the fly).
__device__ __forceinline__ v16h frag_a_f32(const float* tile, int ld) {
  const int lane = threadIdx.x & 31;
  const int row  = lane & 15;
  const int kb   = (lane >> 4) << 3;           // 0 or 8
  const float* p0 = tile + row * ld + kb;
  const float* p1 = p0 + 16;
  v16h f;
#pragma unroll
  for (int i = 0; i < 8; ++i) { f[i] = (_Float16)p0[i]; f[i + 8] = (_Float16)p1[i]; }
  return f;
}

// B-matrix 32x16 f16 (K x N).  Lane holds one column (N = lane&15), K in
// {kb..kb+7, kb+16..kb+23}, kb = (lane>>4)*8.  Source "colT": element
// (k, col) at colT[col*ldk + k] -> two 16B contiguous loads per lane.
__device__ __forceinline__ v16h frag_b_colT(const _Float16* colT, int ldk) {
  const int lane = threadIdx.x & 31;
  const int col  = lane & 15;
  const int kb   = (lane >> 4) << 3;           // 0 or 8
  const v8h lo = *reinterpret_cast<const v8h*>(colT + col * ldk + kb);
  const v8h hi = *reinterpret_cast<const v8h*>(colT + col * ldk + kb + 16);
  v16h f;
#pragma unroll
  for (int i = 0; i < 8; ++i) { f[i] = lo[i]; f[i + 8] = hi[i]; }
  return f;
}

__device__ __forceinline__ float sigmoidf_(float x) {
  return 1.0f / (1.0f + __expf(-x));
}
__device__ __forceinline__ float tanhf_(float x) {
  float e = __expf(-2.0f * x);
  return (1.0f - e) / (1.0f + e);
}

// ---------------- TDM: async 2D tile Global->LDS ----------------------------
// D# per CDNA5 ISA.  2-byte elements; tile 128(x) x 64(y)
// out of a [64][512] f16 plane (row stride 512).  TDM pad feature inserts
// 4 DWORDs (8 halves) after every 64 DWORDs (=one 128-half row) so rows land
// at ld=136 halves in LDS (bank-conflict-free B-fragment reads).
__device__ __forceinline__ void tdm_load_tile_128x64(unsigned lds_addr,
                                                     const _Float16* gsrc) {
  const unsigned long long ga = (unsigned long long)(uintptr_t)gsrc;
  u32x4 g0;
  g0[0] = 1u;                                        // count=1 (valid), user mode
  g0[1] = lds_addr;                                  // LDS byte address
  g0[2] = (unsigned)(ga & 0xFFFFFFFFu);              // global addr [95:64]
  g0[3] = (unsigned)((ga >> 32) & 0x01FFFFFFu)       // global addr [120:96]
        | (2u << 30);                                // type = 2 ("image")
  i32x8 g1;
  g1[0] = (1 << 16)                                  // data_size = 2 bytes
        | (1 << 20)                                  // pad_enable
        | (5 << 22)                                  // pad_interval: 64 DWORDs
        | (3 << 25);                                 // pad_amount: 4 DWORDs
  g1[1] = (512 << 16);                               // tensor_dim0 lo16 = 512
  g1[2] = (64 << 16);                                // dim0 hi=0 | tensor_dim1 lo16 = 64
  g1[3] = (128 << 16);                               // dim1 hi=0 | tile_dim0 = 128
  g1[4] = 64;                                        // tile_dim1 = 64, tile_dim2 = 0
  g1[5] = 512;                                       // tensor_dim0_stride lo32 = 512
  g1[6] = 0;                                         // stride0 hi | stride1 lo
  g1[7] = 0;                                         // stride1 hi
  const i32x4 z = {};
#if __has_include(<hip/amd_detail/amd_gfx1250_TDM.h>)
  const i32x8 z8 = {};
  __builtin_amdgcn_tensor_load_to_lds(g0, g1, z, z, z8, 0);   // clang-23 form
#else
  __builtin_amdgcn_tensor_load_to_lds(g0, g1, z, z, 0);       // ROCm 7.2 form
#endif
}

// ---------------- prep kernels ---------------------------------------------
//  WinT/WoutT : [(e*64 + d)][k]  (64 k-contig halves per column)  = W[k][d*4+e]
//  WrT/WzT/WtT: [d][k] (192 k-contig halves per column)           = W[k][d]
__global__ void prep_weights_kernel(const float* __restrict__ W_in,
                                    const float* __restrict__ W_out,
                                    const float* __restrict__ W_r,
                                    const float* __restrict__ W_z,
                                    const float* __restrict__ W_t,
                                    _Float16* __restrict__ WinT,
                                    _Float16* __restrict__ WoutT,
                                    _Float16* __restrict__ WrT,
                                    _Float16* __restrict__ WzT,
                                    _Float16* __restrict__ WtT) {
  int t = blockIdx.x * 256 + threadIdx.x;
  if (t < 16384) {                   // (e*64+d)*64 + k
    int k = t & 63, d = (t >> 6) & 63, e = t >> 12;
    WinT[t]  = (_Float16)W_in [k * 256 + d * 4 + e];
    WoutT[t] = (_Float16)W_out[k * 256 + d * 4 + e];
  } else if (t < 16384 + 12288) {    // d*192 + k
    int u = t - 16384;
    int k = u % 192, d = u / 192;
    WrT[u] = (_Float16)W_r[k * 64 + d];
    WzT[u] = (_Float16)W_z[k * 64 + d];
    WtT[u] = (_Float16)W_t[k * 64 + d];
  }
}

// adj fp32 [B][N][N][E] -> f16 planes [(b*4+e)][i][j].  134 MB: fits the
// 192 MB L2, so all 5 steps of the dominant GEMM are L2-resident.
__global__ void adj_convert_kernel(const float* __restrict__ adj,
                                   _Float16* __restrict__ adj16) {
  size_t t = (size_t)blockIdx.x * 256 + threadIdx.x;     // over b*i*j
  if (t >= (size_t)B_ * N_ * N_) return;
  int j = (int)(t & 511);
  size_t bi = t >> 9;
  int i = (int)(bi & 511);
  int b = (int)(bi >> 9);
  const float4 v = *reinterpret_cast<const float4*>(adj + t * 4);
  size_t plane = (size_t)N_ * N_;
  size_t o = (size_t)i * N_ + j;
  adj16[(size_t)(b * 4 + 0) * plane + o] = (_Float16)v.x;
  adj16[(size_t)(b * 4 + 1) * plane + o] = (_Float16)v.y;
  adj16[(size_t)(b * 4 + 2) * plane + o] = (_Float16)v.z;
  adj16[(size_t)(b * 4 + 3) * plane + o] = (_Float16)v.w;
}

__global__ void copy_h_kernel(const float* __restrict__ src, float* __restrict__ h) {
  int t = blockIdx.x * 256 + threadIdx.x;
  if (t < BN_ * D_) h[t] = src[t];
}

// ---------------- step kernel 1: lin_in / lin_out ---------------------------
// Per wave: one 16-row tile of h[b], one edge type e; 64 output columns for
// both W_in and W_out (shared A-fragments).  Output stored transposed f16:
// linT[(e*B+b)*D + d][j]  (B-operand layout for the aggregation GEMM).
__global__ void lin_kernel(const float* __restrict__ h,
                           const _Float16* __restrict__ WinT,
                           const float* __restrict__ b_in,
                           const _Float16* __restrict__ WoutT,
                           const float* __restrict__ b_out,
                           _Float16* __restrict__ linT_in,
                           _Float16* __restrict__ linT_out) {
  const int w    = threadIdx.x >> 5;              // 0..7
  const int lane = threadIdx.x & 31;
  const int id   = blockIdx.x * 2 + (w >> 2);     // 0..2047
  const int e    = w & 3;
  const int b    = id >> 5;
  const int jt   = id & 31;

  const float* A = h + ((size_t)b * N_ + jt * 16) * D_;   // 16 x 64, ld 64
  const v16h a0 = frag_a_f32(A, D_);
  const v16h a1 = frag_a_f32(A + 32, D_);

  v8f accI[4] = {}, accO[4] = {};
#pragma unroll
  for (int dt = 0; dt < 4; ++dt) {
    const _Float16* wi = WinT  + (size_t)(e * 64 + dt * 16) * 64;
    const _Float16* wo = WoutT + (size_t)(e * 64 + dt * 16) * 64;
    accI[dt] = WMMA_F16(a0, frag_b_colT(wi, 64), accI[dt]);
    accI[dt] = WMMA_F16(a1, frag_b_colT(wi + 32, 64), accI[dt]);
    accO[dt] = WMMA_F16(a0, frag_b_colT(wo, 64), accO[dt]);
    accO[dt] = WMMA_F16(a1, frag_b_colT(wo + 32, 64), accO[dt]);
  }

  const int col  = lane & 15;
  const int roff = (lane >> 4) << 3;              // 0 or 8
  const int j0   = jt * 16 + roff;
#pragma unroll
  for (int dt = 0; dt < 4; ++dt) {
    const int d  = dt * 16 + col;
    const float bi = b_in [d * 4 + e];
    const float bo = b_out[d * 4 + e];
    v8h vi, vo;
#pragma unroll
    for (int p = 0; p < 8; ++p) {
      vi[p] = (_Float16)(accI[dt][p] + bi);
      vo[p] = (_Float16)(accO[dt][p] + bo);
    }
    const size_t plane = ((size_t)(e * B_ + b) * D_ + d) * N_;
    *reinterpret_cast<v8h*>(linT_in  + plane + j0) = vi;
    *reinterpret_cast<v8h*>(linT_out + plane + j0) = vo;
  }
}

// ---------------- step kernel 2: adjacency aggregation ----------------------
// a_in[b,i,d] = sum_{e,j} adj[b,i,j,e] * lin_in[b,j,d,e]   (same for out).
// Block = 4 waves, 128 i-rows of one b.  Per (e, k-chunk of 128): wave 0
// TDM-stages the lin_in/lin_out chunk ([64 d][128 j] f16, padded to ld=136)
// into LDS, then every wave runs 2 i-tiles x 4 d-tiles x in/out off shared B.
__global__ void __launch_bounds__(128)
agg_kernel(const _Float16* __restrict__ adj16,
           const _Float16* __restrict__ linT_in,
           const _Float16* __restrict__ linT_out,
           float* __restrict__ a_in,
           float* __restrict__ a_out) {
  __shared__ __align__(16) _Float16 sBi[D_ * LDB];   // 17408 B
  __shared__ __align__(16) _Float16 sBo[D_ * LDB];

  const int w    = threadIdx.x >> 5;              // 0..3
  const int lane = threadIdx.x & 31;
  const int b     = blockIdx.x >> 2;              // 256 blocks
  const int strip = blockIdx.x & 3;
  const int i0    = strip * 128 + w * 32;         // wave: rows i0 .. i0+31

  const unsigned lds_i = (unsigned)(uintptr_t)(void*)sBi;
  const unsigned lds_o = (unsigned)(uintptr_t)(void*)sBo;

  v8f accI[2][4] = {{}}, accO[2][4] = {{}};

#pragma unroll 1
  for (int e = 0; e < E_; ++e) {
    const _Float16* Abase = adj16 + ((size_t)(b * 4 + e) * N_ + i0) * N_;
    const _Float16* Bi = linT_in  + (size_t)(e * B_ + b) * D_ * N_;
    const _Float16* Bo = linT_out + (size_t)(e * B_ + b) * D_ * N_;
#pragma unroll 1
    for (int kc = 0; kc < 4; ++kc) {
      __syncthreads();                            // LDS buffers free to refill
      if (threadIdx.x < 32) {                     // wave 0 drives the TDM
        tdm_load_tile_128x64(lds_i, Bi + kc * 128);
        tdm_load_tile_128x64(lds_o, Bo + kc * 128);
        __builtin_amdgcn_s_wait_tensorcnt(0);
      }
      __syncthreads();                            // B chunk visible to all waves
#pragma unroll
      for (int kk = 0; kk < 128; kk += 32) {
        const int k = kc * 128 + kk;
        const v16h a0 = frag_a_f16(Abase + k, N_);
        const v16h a1 = frag_a_f16(Abase + (size_t)16 * N_ + k, N_);
#pragma unroll
        for (int dt = 0; dt < 4; ++dt) {
          const v16h bi = frag_b_colT(sBi + dt * 16 * LDB + kk, LDB);
          const v16h bo = frag_b_colT(sBo + dt * 16 * LDB + kk, LDB);
          accI[0][dt] = WMMA_F16(a0, bi, accI[0][dt]);
          accI[1][dt] = WMMA_F16(a1, bi, accI[1][dt]);
          accO[0][dt] = WMMA_F16(a0, bo, accO[0][dt]);
          accO[1][dt] = WMMA_F16(a1, bo, accO[1][dt]);
        }
      }
    }
  }

  const int col  = lane & 15;
  const int roff = (lane >> 4) << 3;
#pragma unroll
  for (int it = 0; it < 2; ++it) {
#pragma unroll
    for (int dt = 0; dt < 4; ++dt) {
      const int d = dt * 16 + col;
#pragma unroll
      for (int p = 0; p < 8; ++p) {
        const size_t row = (size_t)b * N_ + i0 + it * 16 + roff + p;
        a_in [row * D_ + d] = accI[it][dt][p];
        a_out[row * D_ + d] = accO[it][dt][p];
      }
    }
  }
}

// ---------------- step kernel 3: fused GRU update ---------------------------
// r = sig([a_in,a_out,h]Wr+br), z = sig(...Wz+bz),
// h_hat = tanh([a_in,a_out,r*h]Wt+bt), h = (1-z)h + z*h_hat.
// r*h needs a cross-lane transpose (C-layout -> A-layout): staged via LDS.
__global__ void gru_kernel(const float* __restrict__ a_in,
                           const float* __restrict__ a_out,
                           float* __restrict__ h,
                           const _Float16* __restrict__ WrT, const float* __restrict__ b_r,
                           const _Float16* __restrict__ WzT, const float* __restrict__ b_z,
                           const _Float16* __restrict__ WtT, const float* __restrict__ b_t) {
  __shared__ float lds_rh[4][16 * 64];
  const int w    = threadIdx.x >> 5;              // 0..3
  const int lane = threadIdx.x & 31;
  const int rt   = blockIdx.x * 4 + w;            // 16-row tile of BN rows
  const size_t row0 = (size_t)rt * 16;

  const float* Ain  = a_in  + row0 * D_;
  const float* Aout = a_out + row0 * D_;
  const float* H    = h     + row0 * D_;

  v16h af[6];
  af[0] = frag_a_f32(Ain, D_);       af[1] = frag_a_f32(Ain + 32, D_);
  af[2] = frag_a_f32(Aout, D_);      af[3] = frag_a_f32(Aout + 32, D_);
  af[4] = frag_a_f32(H, D_);         af[5] = frag_a_f32(H + 32, D_);

  v8f racc[4] = {}, zacc[4] = {};
#pragma unroll
  for (int kf = 0; kf < 6; ++kf) {
#pragma unroll
    for (int dt = 0; dt < 4; ++dt) {
      racc[dt] = WMMA_F16(af[kf], frag_b_colT(WrT + dt * 16 * 192 + kf * 32, 192), racc[dt]);
      zacc[dt] = WMMA_F16(af[kf], frag_b_colT(WzT + dt * 16 * 192 + kf * 32, 192), zacc[dt]);
    }
  }

  const int col  = lane & 15;
  const int roff = (lane >> 4) << 3;
  float hv[4][8];
  float* my = &lds_rh[w][0];
#pragma unroll
  for (int dt = 0; dt < 4; ++dt) {
    const int d = dt * 16 + col;
    const float rb = b_r[d], zb = b_z[d];
#pragma unroll
    for (int p = 0; p < 8; ++p) {
      const int m = roff + p;
      const float hval = H[(size_t)m * D_ + d];
      hv[dt][p] = hval;
      const float r = sigmoidf_(racc[dt][p] + rb);
      zacc[dt][p] = sigmoidf_(zacc[dt][p] + zb);     // keep z in-register
      my[m * 64 + d] = r * hval;                     // C-layout -> LDS
    }
  }
  __syncthreads();

  af[4] = frag_a_f32(my, 64);                        // LDS -> A-layout
  af[5] = frag_a_f32(my + 32, 64);

  v8f tacc[4] = {};
#pragma unroll
  for (int kf = 0; kf < 6; ++kf) {
#pragma unroll
    for (int dt = 0; dt < 4; ++dt) {
      tacc[dt] = WMMA_F16(af[kf], frag_b_colT(WtT + dt * 16 * 192 + kf * 32, 192), tacc[dt]);
    }
  }

#pragma unroll
  for (int dt = 0; dt < 4; ++dt) {
    const int d = dt * 16 + col;
    const float tb = b_t[d];
#pragma unroll
    for (int p = 0; p < 8; ++p) {
      const float hh = tanhf_(tacc[dt][p] + tb);
      const float z  = zacc[dt][p];
      const float hn = (1.0f - z) * hv[dt][p] + z * hh;
      h[(row0 + roff + p) * D_ + d] = hn;
    }
  }
}

// ---------------- output head (algebraically folded) -----------------------
// sum_d([h,ann]W_o + b_o) = [h,ann] . rowsum(W_o) + sum(b_o)
__global__ void wsum_kernel(const float* __restrict__ W_o,
                            const float* __restrict__ b_o,
                            float* __restrict__ wsum) {
  int t = threadIdx.x;
  if (t < D_ + AD_) {
    float s = 0.f;
    for (int d = 0; d < D_; ++d) s += W_o[t * D_ + d];
    wsum[t] = s;
  } else if (t == D_ + AD_) {
    float s = 0.f;
    for (int d = 0; d < D_; ++d) s += b_o[d];
    wsum[D_ + AD_] = s;
  }
}

__global__ void head_kernel(const float* __restrict__ h,
                            const float* __restrict__ ann,
                            const float* __restrict__ wsum,
                            float* __restrict__ out) {
  int row = blockIdx.x * 256 + threadIdx.x;
  if (row >= BN_) return;
  float s = wsum[D_ + AD_];
#pragma unroll 8
  for (int k = 0; k < D_; ++k)  s += h[(size_t)row * D_ + k] * wsum[k];
#pragma unroll
  for (int k = 0; k < AD_; ++k) s += ann[(size_t)row * AD_ + k] * wsum[D_ + k];
  out[row] = s;
}

// ---------------------------------------------------------------------------
extern "C" void kernel_launch(void* const* d_in, const int* in_sizes, int n_in,
                              void* d_out, int out_size, void* d_ws, size_t ws_size,
                              hipStream_t stream) {
  (void)in_sizes; (void)n_in; (void)out_size; (void)ws_size;
  const float* init_state = (const float*)d_in[0];
  const float* annotation = (const float*)d_in[1];
  const float* adj        = (const float*)d_in[2];
  const float* W_in  = (const float*)d_in[3];
  const float* b_in  = (const float*)d_in[4];
  const float* W_out = (const float*)d_in[5];
  const float* b_out = (const float*)d_in[6];
  const float* W_r = (const float*)d_in[7];   const float* b_r = (const float*)d_in[8];
  const float* W_z = (const float*)d_in[9];   const float* b_z = (const float*)d_in[10];
  const float* W_t = (const float*)d_in[11];  const float* b_t = (const float*)d_in[12];
  const float* W_o = (const float*)d_in[13];  const float* b_o = (const float*)d_in[14];
  float* out = (float*)d_out;

  char* base = (char*)d_ws;
  size_t off = 0;
  auto take = [&](size_t bytes) -> char* {
    char* p = base + off;
    off = (off + bytes + 255) & ~(size_t)255;
    return p;
  };

  float*     h        = (float*)take((size_t)BN_ * D_ * 4);
  float*     a_in     = (float*)take((size_t)BN_ * D_ * 4);
  float*     a_out    = (float*)take((size_t)BN_ * D_ * 4);
  _Float16*  linT_in  = (_Float16*)take((size_t)E_ * B_ * D_ * N_ * 2);
  _Float16*  linT_out = (_Float16*)take((size_t)E_ * B_ * D_ * N_ * 2);
  _Float16*  WinT     = (_Float16*)take((size_t)D_ * E_ * D_ * 2);
  _Float16*  WoutT    = (_Float16*)take((size_t)D_ * E_ * D_ * 2);
  _Float16*  WrT      = (_Float16*)take((size_t)D_ * 3 * D_ * 2);
  _Float16*  WzT      = (_Float16*)take((size_t)D_ * 3 * D_ * 2);
  _Float16*  WtT      = (_Float16*)take((size_t)D_ * 3 * D_ * 2);
  float*     wsum     = (float*)take((D_ + AD_ + 1) * 4);
  _Float16*  adj16    = (_Float16*)take((size_t)B_ * E_ * N_ * N_ * 2);

  prep_weights_kernel<<<(16384 + 12288 + 255) / 256, 256, 0, stream>>>(
      W_in, W_out, W_r, W_z, W_t, WinT, WoutT, WrT, WzT, WtT);
  adj_convert_kernel<<<(B_ * N_ * N_) / 256, 256, 0, stream>>>(adj, adj16);
  copy_h_kernel<<<(BN_ * D_) / 256, 256, 0, stream>>>(init_state, h);

  for (int step = 0; step < 5; ++step) {
    lin_kernel<<<1024, 256, 0, stream>>>(h, WinT, b_in, WoutT, b_out, linT_in, linT_out);
    agg_kernel<<<256, 128, 0, stream>>>(adj16, linT_in, linT_out, a_in, a_out);
    gru_kernel<<<512, 128, 0, stream>>>(a_in, a_out, h, WrT, b_r, WzT, b_z, WtT, b_t);
  }

  wsum_kernel<<<1, 128, 0, stream>>>(W_o, b_o, wsum);
  head_kernel<<<BN_ / 256, 256, 0, stream>>>(h, annotation, wsum, out);
}
